// ScaledDotProductAttention_71614284693937
// MI455X (gfx1250) — compile-verified, matrix operand order checked by reading, not run on hardware
//
#include <hip/hip_runtime.h>

#define B_   8
#define LQ_  2048
#define LK_  2048
#define D_   512

typedef __attribute__((ext_vector_type(16))) __bf16 v16bf;
typedef __attribute__((ext_vector_type(8)))  float  v8f;

union BFrag {
    v16bf  v;
    __bf16 h[16];
    uint4  q[2];
};

union BPack2 {
    __bf16       h[2];
    unsigned int w;
};

union BF1 {
    __bf16         h;
    unsigned short u;
};

__device__ __forceinline__ unsigned short bfbits(float x) {
    BF1 t; t.h = (__bf16)x; return t.u;
}

// ---------------- Pre-pass 1: K f32 -> bf16, row-major ----------------
__global__ void __launch_bounds__(256)
convert_bf16_kernel(const float* __restrict__ src, __bf16* __restrict__ dst) {
    const int i = blockIdx.x * 256 + threadIdx.x;   // one float4 per thread
    const float4 v = ((const float4*)src)[i];
    BPack2 a, b;
    a.h[0] = (__bf16)v.x; a.h[1] = (__bf16)v.y;
    b.h[0] = (__bf16)v.z; b.h[1] = (__bf16)v.w;
    uint2 w; w.x = a.w; w.y = b.w;
    ((uint2*)dst)[i] = w;
}

// ---------------- Pre-pass 2: V f32 -> bf16, transposed VT[b][d][k] ----------------
__global__ void __launch_bounds__(256)
transpose_v_bf16_kernel(const float* __restrict__ Vf, __bf16* __restrict__ VT) {
    __shared__ unsigned short tile[64][66];          // +2 pad: spread LDS banks
    const int dblk = blockIdx.x % (D_ / 64);
    const int kblk = (blockIdx.x / (D_ / 64)) % (LK_ / 64);
    const int b    = blockIdx.x / ((D_ / 64) * (LK_ / 64));
    const int k0 = kblk * 64, d0 = dblk * 64;

    const float* src = Vf + ((size_t)b * LK_ + k0) * D_ + d0;
    for (int i = threadIdx.x; i < 64 * 16; i += 256) {   // coalesced float4 reads
        const int k = i >> 4, d4 = (i & 15) * 4;
        const float4 v4 = *(const float4*)(src + (size_t)k * D_ + d4);
        tile[d4 + 0][k] = bfbits(v4.x);
        tile[d4 + 1][k] = bfbits(v4.y);
        tile[d4 + 2][k] = bfbits(v4.z);
        tile[d4 + 3][k] = bfbits(v4.w);
    }
    __syncthreads();
    __bf16* dst = VT + ((size_t)b * D_ + d0) * LK_ + k0;
    for (int i = threadIdx.x; i < 64 * 32; i += 256) {   // coalesced dword writes
        const int d = i >> 5, kp = (i & 31) * 2;
        const unsigned int w =
            (unsigned int)tile[d][kp] | ((unsigned int)tile[d][kp + 1] << 16);
        *(unsigned int*)((__bf16*)dst + (size_t)d * LK_ + kp) = w;
    }
}

// ---------------- Main fused attention kernel (M = 32 queries / block) ----------------
template <bool PRE>
__global__ void __launch_bounds__(512)
attn_fwd_kernel(const float* __restrict__ Q,
                const float* __restrict__ Kf,
                const float* __restrict__ Vf,
                const __bf16* __restrict__ K16,
                const __bf16* __restrict__ VT,
                const int*   __restrict__ mask,
                const int*   __restrict__ revp,
                float* __restrict__ outO,
                float* __restrict__ outW) {
    extern __shared__ char smem[];
    __bf16* q_lds = (__bf16*)smem;                    // [32][512] bf16, 32 KB (reused as V slab in fallback)
    float*  s_lds = (float*)(smem + 32 * D_ * 2);     // [32][2048] f32, 256 KB (P packs in place)

    const int tid    = threadIdx.x;
    const int lane   = tid & 31;
    const int wave   = tid >> 5;                      // 16 waves
    const int laneHi = lane >> 4;
    const int lane15 = lane & 15;
    const int l2     = laneHi ? 8 : 0;

    const int b  = blockIdx.x / (LQ_ / 32);
    const int q0 = (blockIdx.x % (LQ_ / 32)) * 32;

    const float NEG_INF = -__builtin_inff();
    const float scl = ((*revp) ? -1.0f : 1.0f) * 0.044194173824159216f; // +-1/sqrt(512)

    // ---- Phase 0: stage Q tile (fp32 -> bf16) ----
    const float* qbase = Q + ((size_t)b * LQ_ + q0) * D_;
    for (int i = tid; i < 32 * D_; i += 512)
        q_lds[i] = (__bf16)qbase[i];
    __syncthreads();

    // ---- Phase 1: S = sign*(Q K^T)/temper, masked (2 key tiles share one A-frag) ----
    const int* mbase = mask + (size_t)b * LK_;
    for (int u = wave; u < 2 * (LK_ / 32); u += 16) {   // 128 units / 16 waves
        const int qs  = u & 1;
        const int k0t = (u >> 1) * 32;                  // 32 keys per unit
        v8f acc0 = {}, acc1 = {};
        const __bf16* qrow = q_lds + (size_t)(qs * 16 + lane15) * D_;
#pragma unroll 4
        for (int d0 = 0; d0 < D_; d0 += 32) {
            const int lo = d0 + l2;
            BFrag a, b0, b1;
            a.q[0] = *(const uint4*)(qrow + lo);
            a.q[1] = *(const uint4*)(qrow + lo + 16);
            if constexpr (PRE) {
                const __bf16* krow0 = K16 + ((size_t)b * LK_ + k0t + lane15) * D_;
                const __bf16* krow1 = krow0 + (size_t)16 * D_;
                b0.q[0] = *(const uint4*)(krow0 + lo);
                b0.q[1] = *(const uint4*)(krow0 + lo + 16);
                b1.q[0] = *(const uint4*)(krow1 + lo);
                b1.q[1] = *(const uint4*)(krow1 + lo + 16);
            } else {
                const float* krow0 = Kf + ((size_t)b * LK_ + k0t + lane15) * D_;
                const float* krow1 = krow0 + (size_t)16 * D_;
                float kf[16], kg[16];
                *(float4*)(kf + 0)  = *(const float4*)(krow0 + lo);
                *(float4*)(kf + 4)  = *(const float4*)(krow0 + lo + 4);
                *(float4*)(kf + 8)  = *(const float4*)(krow0 + lo + 16);
                *(float4*)(kf + 12) = *(const float4*)(krow0 + lo + 20);
                *(float4*)(kg + 0)  = *(const float4*)(krow1 + lo);
                *(float4*)(kg + 4)  = *(const float4*)(krow1 + lo + 4);
                *(float4*)(kg + 8)  = *(const float4*)(krow1 + lo + 16);
                *(float4*)(kg + 12) = *(const float4*)(krow1 + lo + 20);
#pragma unroll
                for (int e = 0; e < 16; ++e) { b0.h[e] = (__bf16)kf[e]; b1.h[e] = (__bf16)kg[e]; }
            }
            acc0 = __builtin_amdgcn_wmma_f32_16x16x32_bf16(
                false, a.v, false, b0.v, (short)0, acc0, false, false);
            acc1 = __builtin_amdgcn_wmma_f32_16x16x32_bf16(
                false, a.v, false, b1.v, (short)0, acc1, false, false);
        }
        const int mk0 = mbase[k0t + lane15];
        const int mk1 = mbase[k0t + 16 + lane15];
        float* srow0 = s_lds + (size_t)(qs * 16) * LK_ + k0t + lane15;
        float* srow1 = srow0 + 16;
#pragma unroll
        for (int j = 0; j < 8; ++j) {
            const int r = j + (laneHi ? 8 : 0);
            srow0[(size_t)r * LK_] = mk0 ? acc0[j] * scl : NEG_INF;
            srow1[(size_t)r * LK_] = mk1 ? acc1[j] * scl : NEG_INF;
        }
    }
    __syncthreads();

    // ---- Phase 2: softmax (fused exp); f32 weights to HBM, bf16 P packed in place ----
    float* wbase = outW + ((size_t)b * LQ_ + q0) * LK_;
#pragma unroll
    for (int rr = 0; rr < 2; ++rr) {
        const int r = wave * 2 + rr;                   // 16 waves x 2 rows = 32 rows
        float* srow = s_lds + (size_t)r * LK_;
        float m = NEG_INF;
        for (int kk = lane; kk < LK_; kk += 32) m = fmaxf(m, srow[kk]);
#pragma unroll
        for (int off = 16; off > 0; off >>= 1) m = fmaxf(m, __shfl_xor(m, off, 32));
        const bool live = (m > NEG_INF);
        float s = 0.0f;
        if (live)
            for (int kk = lane; kk < LK_; kk += 32) {  // exp once, written back in place
                const float e = __expf(srow[kk] - m);
                srow[kk] = e;
                s += e;
            }
#pragma unroll
        for (int off = 16; off > 0; off >>= 1) s += __shfl_xor(s, off, 32);
        const float inv = (live && s > 0.0f) ? (1.0f / s) : 0.0f;
        float* wrow = wbase + (size_t)r * LK_;
        // pack P pairs into the front of the same f32 row (same-type stores; write
        // range [32i,32i+31] never overlaps read range [64j,64j+63] for j>i)
        for (int j = lane; j < LK_ / 2; j += 32) {
            const float p0 = live ? srow[2 * j]     * inv : 0.0f;
            const float p1 = live ? srow[2 * j + 1] * inv : 0.0f;
            BPack2 pk; pk.h[0] = (__bf16)p0; pk.h[1] = (__bf16)p1;
            float2 w2; w2.x = p0; w2.y = p1;
            *(float2*)(wrow + 2 * j) = w2;             // stream f32 weights
            srow[j] = __uint_as_float(pk.w);           // bf16 P, contiguous in row front
        }
    }
    __syncthreads();

    // ---- Phase 3: O = P * V, 2x2 register blocking (each frag feeds 2 WMMAs) ----
    float* obase = outO + ((size_t)b * LQ_ + q0) * D_;
    v8f acc[4] = {v8f{}, v8f{}, v8f{}, v8f{}};         // [qs*2 + h]
    const __bf16* prow0 = (const __bf16*)(s_lds + (size_t)lane15 * LK_);
    const __bf16* prow1 = (const __bf16*)(s_lds + (size_t)(16 + lane15) * LK_);

    if constexpr (PRE) {
        const __bf16* vtrow0 = VT + ((size_t)b * D_ + wave * 16 + lane15) * LK_;
        const __bf16* vtrow1 = VT + ((size_t)b * D_ + (wave + 16) * 16 + lane15) * LK_;
#pragma unroll 2
        for (int k0 = 0; k0 < LK_; k0 += 32) {
            const int lo = k0 + l2;
            BFrag a0, a1, b0, b1;
            a0.q[0] = *(const uint4*)(prow0 + lo);
            a0.q[1] = *(const uint4*)(prow0 + lo + 16);
            a1.q[0] = *(const uint4*)(prow1 + lo);
            a1.q[1] = *(const uint4*)(prow1 + lo + 16);
            b0.q[0] = *(const uint4*)(vtrow0 + lo);
            b0.q[1] = *(const uint4*)(vtrow0 + lo + 16);
            b1.q[0] = *(const uint4*)(vtrow1 + lo);
            b1.q[1] = *(const uint4*)(vtrow1 + lo + 16);
            if (k0 + 32 < LK_) {
                __builtin_prefetch(vtrow0 + lo + 32, 0, 1);
                __builtin_prefetch(vtrow1 + lo + 32, 0, 1);
            }
            acc[0] = __builtin_amdgcn_wmma_f32_16x16x32_bf16(
                false, a0.v, false, b0.v, (short)0, acc[0], false, false);
            acc[1] = __builtin_amdgcn_wmma_f32_16x16x32_bf16(
                false, a0.v, false, b1.v, (short)0, acc[1], false, false);
            acc[2] = __builtin_amdgcn_wmma_f32_16x16x32_bf16(
                false, a1.v, false, b0.v, (short)0, acc[2], false, false);
            acc[3] = __builtin_amdgcn_wmma_f32_16x16x32_bf16(
                false, a1.v, false, b1.v, (short)0, acc[3], false, false);
        }
    } else {
        // Fallback: cooperative transposed V slab in LDS (reuses dead q_lds region)
        __bf16* v_slab = q_lds;                        // [512][32] bf16, 32 KB
        const float* vbase = Vf + (size_t)b * LK_ * D_;
        for (int k0 = 0; k0 < LK_; k0 += 32) {
            __syncthreads();
            for (int it = tid; it < 16 * (D_ / 4); it += 512) {
                const int kk2 = (it & 15) * 2;
                const int d4  = (it >> 4) * 4;
                const float4 ra = *(const float4*)(vbase + (size_t)(k0 + kk2) * D_ + d4);
                const float4 rb = *(const float4*)(vbase + (size_t)(k0 + kk2 + 1) * D_ + d4);
                if (k0 + 32 < LK_)
                    __builtin_prefetch(vbase + (size_t)(k0 + 32 + kk2) * D_ + d4, 0, 1);
                const float fa[4] = {ra.x, ra.y, ra.z, ra.w};
                const float fb[4] = {rb.x, rb.y, rb.z, rb.w};
#pragma unroll
                for (int j = 0; j < 4; ++j) {
                    BPack2 p2; p2.h[0] = (__bf16)fa[j]; p2.h[1] = (__bf16)fb[j];
                    *(unsigned int*)(v_slab + (size_t)(d4 + j) * 32 + kk2) = p2.w;
                }
            }
            __syncthreads();
            const int lo = k0 + l2;
            const __bf16* vrow0 = v_slab + (size_t)(wave * 16 + lane15) * 32;
            const __bf16* vrow1 = v_slab + (size_t)((wave + 16) * 16 + lane15) * 32;
            BFrag a0, a1, b0, b1;
            a0.q[0] = *(const uint4*)(prow0 + lo);
            a0.q[1] = *(const uint4*)(prow0 + lo + 16);
            a1.q[0] = *(const uint4*)(prow1 + lo);
            a1.q[1] = *(const uint4*)(prow1 + lo + 16);
            b0.q[0] = *(const uint4*)(vrow0 + l2);
            b0.q[1] = *(const uint4*)(vrow0 + l2 + 16);
            b1.q[0] = *(const uint4*)(vrow1 + l2);
            b1.q[1] = *(const uint4*)(vrow1 + l2 + 16);
            acc[0] = __builtin_amdgcn_wmma_f32_16x16x32_bf16(
                false, a0.v, false, b0.v, (short)0, acc[0], false, false);
            acc[1] = __builtin_amdgcn_wmma_f32_16x16x32_bf16(
                false, a0.v, false, b1.v, (short)0, acc[1], false, false);
            acc[2] = __builtin_amdgcn_wmma_f32_16x16x32_bf16(
                false, a1.v, false, b0.v, (short)0, acc[2], false, false);
            acc[3] = __builtin_amdgcn_wmma_f32_16x16x32_bf16(
                false, a1.v, false, b1.v, (short)0, acc[3], false, false);
        }
    }

#pragma unroll
    for (int qs = 0; qs < 2; ++qs)
#pragma unroll
        for (int h = 0; h < 2; ++h) {
            const int n0 = (wave + 16 * h) * 16;
#pragma unroll
            for (int j = 0; j < 8; ++j) {
                const int r = qs * 16 + j + (laneHi ? 8 : 0);
                obase[(size_t)r * D_ + n0 + lane15] = acc[qs * 2 + h][j];
            }
        }
}

extern "C" void kernel_launch(void* const* d_in, const int* in_sizes, int n_in,
                              void* d_out, int out_size, void* d_ws, size_t ws_size,
                              hipStream_t stream) {
    (void)in_sizes; (void)n_in; (void)out_size;
    const float* q   = (const float*)d_in[0];
    const float* k   = (const float*)d_in[1];
    const float* v   = (const float*)d_in[2];
    const int*   msk = (const int*)d_in[3];
    const int*   rev = (const int*)d_in[4];
    float* outO = (float*)d_out;                       // [B,LQ,D]
    float* outW = outO + (size_t)B_ * LQ_ * D_;        // [B,LQ,LK]

    const size_t nKV  = (size_t)B_ * LK_ * D_;         // elements in K (== V)
    const size_t need = 2 * nKV * sizeof(__bf16);      // bf16 K + transposed bf16 V
    const size_t shmem = (size_t)32 * D_ * 2           // Q bf16 / V slab
                       + (size_t)32 * LK_ * 4;         // scores f32 (P packs in place)
    dim3 grid(B_ * (LQ_ / 32));
    dim3 block(512);

    if (ws_size >= need) {
        __bf16* K16 = (__bf16*)d_ws;
        __bf16* VT  = K16 + nKV;
        convert_bf16_kernel<<<dim3((unsigned)(nKV / 4 / 256)), dim3(256), 0, stream>>>(k, K16);
        transpose_v_bf16_kernel<<<dim3(B_ * (LK_ / 64) * (D_ / 64)), dim3(256), 0, stream>>>(v, VT);
        attn_fwd_kernel<true><<<grid, block, shmem, stream>>>(q, k, v, K16, VT, msk, rev, outO, outW);
    } else {
        attn_fwd_kernel<false><<<grid, block, shmem, stream>>>(q, k, v, (const __bf16*)d_ws,
                                                               (const __bf16*)d_ws, msk, rev, outO, outW);
    }
}
